// FSPLoss_85555748537092
// MI455X (gfx1250) — compile-verified, matrix operand order checked by reading, not run on hardware
//
#include <hip/hip_runtime.h>
#include <hip/hip_bf16.h>

typedef __attribute__((ext_vector_type(16))) __bf16          v16bf;
typedef __attribute__((ext_vector_type(16))) unsigned short  v16us;
typedef __attribute__((ext_vector_type(8)))  float           v8f;

#define B_SZ 16
#define C_SZ 512
#define T_SZ 4096

__device__ __forceinline__ v16bf ld_frag(const unsigned short* p) {
  v16us u = *reinterpret_cast<const v16us*>(p);
  return __builtin_bit_cast(v16bf, u);
}

// Split a pair of fp32 into packed bf16 hi (truncation) and packed bf16 lo
// (exact Dekker remainder, truncated). One v_perm_b32 packs each pair.
__device__ __forceinline__ void cvt_pair(float f0, float f1,
                                         unsigned& hp, unsigned& lp) {
  unsigned x0 = __float_as_uint(f0);
  unsigned x1 = __float_as_uint(f1);
  hp = __builtin_amdgcn_perm(x1, x0, 0x07060302u);   // {x1[31:16], x0[31:16]}
  float h0 = __uint_as_float(x0 & 0xFFFF0000u);
  float h1 = __uint_as_float(x1 & 0xFFFF0000u);
  float l0 = f0 - h0;
  float l1 = f1 - h1;
  lp = __builtin_amdgcn_perm(__float_as_uint(l1), __float_as_uint(l0), 0x07060302u);
}

// One workgroup = 16 waves = 128x128 output tile of (s2*s1^T - t2*t1^T) per batch.
// Wave w: i-tile pair 2*(w>>2), j-tile pair 2*(w&3); 24 WMMA per k-chunk per wave.
// Split-precision bf16: acc += Ahi*Bhi + Ahi*Blo + Alo*Bhi (fp32 accumulate).
__global__ __launch_bounds__(512) void fsp_wmma_kernel(
    const float* __restrict__ s1, const float* __restrict__ s2,
    const float* __restrict__ t1, const float* __restrict__ t2,
    float* __restrict__ partials)
{
  // K order per row: [0..7, 16..23, 8..15, 24..31] so each lane reads 32B contiguous.
  // Double buffered: one barrier per k-chunk.
  __shared__ __align__(64) unsigned short lds_hi[2][4][128][32];  // 64 KB
  __shared__ __align__(64) unsigned short lds_lo[2][4][128][32];  // 64 KB

  const int tid  = threadIdx.x;
  const int lane = tid & 31;
  const int w    = tid >> 5;          // 0..15
  const int m    = lane & 15;
  const int half = lane >> 4;
  const int wi   = w >> 2;            // 0..3 : i tile pair
  const int wj   = w & 3;             // 0..3 : j tile pair

  const int b  = blockIdx.z;
  const int i0 = blockIdx.y * 128;
  const int j0 = blockIdx.x * 128;

  const float* base[4];
  base[0] = s2 + ((size_t)b * C_SZ + i0) * T_SZ;  // A rows (top), student
  base[1] = s1 + ((size_t)b * C_SZ + j0) * T_SZ;  // B rows (bot), student
  base[2] = t2 + ((size_t)b * C_SZ + i0) * T_SZ;  // A rows (top), teacher
  base[3] = t1 + ((size_t)b * C_SZ + j0) * T_SZ;  // B rows (bot), teacher

  // staging: 512 threads cover 128 rows x 32 k per matrix (8 floats/thread/matrix)
  const int sr    = tid >> 2;         // row 0..127
  const int seg   = tid & 3;          // 8-wide k segment
  const int kk    = seg << 3;         // 0,8,16,24
  const int permk = (seg == 1) ? (kk + 8) : ((seg == 2) ? (kk - 8) : kk);

  auto stage = [&](int kbase, int buf) {
    #pragma unroll
    for (int mat = 0; mat < 4; ++mat) {
      const float* src = base[mat] + (size_t)sr * T_SZ + (kbase + kk);
      float4 f0 = *reinterpret_cast<const float4*>(src);
      float4 f1 = *reinterpret_cast<const float4*>(src + 4);
      if (kbase + 32 < T_SZ) __builtin_prefetch(src + 32, 0, 0);  // global_prefetch_b8
      unsigned hp0, lp0, hp1, lp1, hp2, lp2, hp3, lp3;
      cvt_pair(f0.x, f0.y, hp0, lp0);
      cvt_pair(f0.z, f0.w, hp1, lp1);
      cvt_pair(f1.x, f1.y, hp2, lp2);
      cvt_pair(f1.z, f1.w, hp3, lp3);
      *reinterpret_cast<uint4*>(&lds_hi[buf][mat][sr][permk]) =
          make_uint4(hp0, hp1, hp2, hp3);
      *reinterpret_cast<uint4*>(&lds_lo[buf][mat][sr][permk]) =
          make_uint4(lp0, lp1, lp2, lp3);
    }
  };

  v8f acc_s00 = {}, acc_s01 = {}, acc_s10 = {}, acc_s11 = {};
  v8f acc_t00 = {}, acc_t01 = {}, acc_t10 = {}, acc_t11 = {};

  const int a0row = (2 * wi + 0) * 16 + m;
  const int a1row = (2 * wi + 1) * 16 + m;
  const int b0row = (2 * wj + 0) * 16 + m;
  const int b1row = (2 * wj + 1) * 16 + m;
  const int kc    = half * 16;

  // 12 WMMA for one (A-mat, B-mat) pair into four accumulators
  auto gemm_phase = [&](int buf, int matA, int matB,
                        v8f& a00, v8f& a01, v8f& a10, v8f& a11) {
    v16bf a0h = ld_frag(&lds_hi[buf][matA][a0row][kc]);
    v16bf a0l = ld_frag(&lds_lo[buf][matA][a0row][kc]);
    v16bf a1h = ld_frag(&lds_hi[buf][matA][a1row][kc]);
    v16bf a1l = ld_frag(&lds_lo[buf][matA][a1row][kc]);
    v16bf b0h = ld_frag(&lds_hi[buf][matB][b0row][kc]);
    v16bf b0l = ld_frag(&lds_lo[buf][matB][b0row][kc]);
    v16bf b1h = ld_frag(&lds_hi[buf][matB][b1row][kc]);
    v16bf b1l = ld_frag(&lds_lo[buf][matB][b1row][kc]);
    a00 = __builtin_amdgcn_wmma_f32_16x16x32_bf16(false, a0h, false, b0h, (short)0, a00, false, false);
    a00 = __builtin_amdgcn_wmma_f32_16x16x32_bf16(false, a0h, false, b0l, (short)0, a00, false, false);
    a00 = __builtin_amdgcn_wmma_f32_16x16x32_bf16(false, a0l, false, b0h, (short)0, a00, false, false);
    a01 = __builtin_amdgcn_wmma_f32_16x16x32_bf16(false, a0h, false, b1h, (short)0, a01, false, false);
    a01 = __builtin_amdgcn_wmma_f32_16x16x32_bf16(false, a0h, false, b1l, (short)0, a01, false, false);
    a01 = __builtin_amdgcn_wmma_f32_16x16x32_bf16(false, a0l, false, b1h, (short)0, a01, false, false);
    a10 = __builtin_amdgcn_wmma_f32_16x16x32_bf16(false, a1h, false, b0h, (short)0, a10, false, false);
    a10 = __builtin_amdgcn_wmma_f32_16x16x32_bf16(false, a1h, false, b0l, (short)0, a10, false, false);
    a10 = __builtin_amdgcn_wmma_f32_16x16x32_bf16(false, a1l, false, b0h, (short)0, a10, false, false);
    a11 = __builtin_amdgcn_wmma_f32_16x16x32_bf16(false, a1h, false, b1h, (short)0, a11, false, false);
    a11 = __builtin_amdgcn_wmma_f32_16x16x32_bf16(false, a1h, false, b1l, (short)0, a11, false, false);
    a11 = __builtin_amdgcn_wmma_f32_16x16x32_bf16(false, a1l, false, b1h, (short)0, a11, false, false);
  };

  stage(0, 0);
  __syncthreads();

  int ibuf = 0;
  for (int k0 = 0; k0 < T_SZ; k0 += 32, ibuf ^= 1) {
    if (k0 + 32 < T_SZ) stage(k0 + 32, ibuf ^ 1);        // fill other buffer
    gemm_phase(ibuf, 0, 1, acc_s00, acc_s01, acc_s10, acc_s11);  // student
    gemm_phase(ibuf, 2, 3, acc_t00, acc_t01, acc_t10, acc_t11);  // teacher
    __syncthreads();  // next iter reads what we just staged; frees current buffer
  }

  // ---- epilogue: sum ((s-t)/T)^2 over this wave's four 16x16 tiles ----
  const float invT = 1.0f / (float)T_SZ;
  float sumv = 0.0f;
  #pragma unroll
  for (int e = 0; e < 8; ++e) {
    float d0 = (acc_s00[e] - acc_t00[e]) * invT;
    float d1 = (acc_s01[e] - acc_t01[e]) * invT;
    float d2 = (acc_s10[e] - acc_t10[e]) * invT;
    float d3 = (acc_s11[e] - acc_t11[e]) * invT;
    sumv += d0 * d0 + d1 * d1 + d2 * d2 + d3 * d3;
  }
  #pragma unroll
  for (int off = 16; off > 0; off >>= 1)
    sumv += __shfl_xor(sumv, off, 32);   // wave32 reduction
  if (lane == 0) {
    int bid = (blockIdx.z * gridDim.y + blockIdx.y) * gridDim.x + blockIdx.x;
    partials[bid * 16 + w] = sumv;
  }
}

// Deterministic final reduction of per-wave partials.
__global__ __launch_bounds__(256) void fsp_reduce_kernel(
    const float* __restrict__ partials, float* __restrict__ out, int n)
{
  __shared__ float sm[256];
  float s = 0.0f;
  for (int i = threadIdx.x; i < n; i += 256) s += partials[i];
  sm[threadIdx.x] = s;
  __syncthreads();
  for (int st = 128; st > 0; st >>= 1) {
    if ((int)threadIdx.x < st) sm[threadIdx.x] += sm[threadIdx.x + st];
    __syncthreads();
  }
  if (threadIdx.x == 0)
    out[0] = sm[0] * (1.0f / ((float)B_SZ * (float)C_SZ * (float)C_SZ));
}

extern "C" void kernel_launch(void* const* d_in, const int* in_sizes, int n_in,
                              void* d_out, int out_size, void* d_ws, size_t ws_size,
                              hipStream_t stream) {
  const float* s1 = (const float*)d_in[0];
  const float* s2 = (const float*)d_in[1];
  const float* t1 = (const float*)d_in[2];
  const float* t2 = (const float*)d_in[3];
  float* partials = (float*)d_ws;          // 4*4*16*16 = 4096 floats = 16 KB

  dim3 grid(C_SZ / 128, C_SZ / 128, B_SZ); // (4, 4, 16); batch slowest for L2 reuse
  dim3 block(512);
  fsp_wmma_kernel<<<grid, block, 0, stream>>>(s1, s2, t1, t2, partials);

  const int nPartials = (C_SZ / 128) * (C_SZ / 128) * B_SZ * 16;
  fsp_reduce_kernel<<<1, 256, 0, stream>>>(partials, (float*)d_out, nPartials);
}
